// DGI_89103391523491
// MI455X (gfx1250) — compile-verified, matrix operand order checked by reading, not run on hardware
//
#include <hip/hip_runtime.h>
#include <hip/hip_bf16.h>

// ---------- problem constants ----------
constexpr int NN   = 50000;   // nodes
constexpr int NE   = 800000;  // edges
constexpr int NG   = 64;      // graphs
constexpr int DIN  = 128;
constexpr int DH   = 256;

// ---------- vector types for WMMA ----------
typedef __attribute__((ext_vector_type(16))) _Float16 v16h;
typedef __attribute__((ext_vector_type(8)))  _Float16 v8h;
typedef __attribute__((ext_vector_type(8)))  float    v8f;

// ============================================================
// utility kernels
// ============================================================
__global__ void fill_f32(float* __restrict__ p, float v, long long n) {
    long long i = (long long)blockIdx.x * blockDim.x + threadIdx.x;
    long long stride = (long long)gridDim.x * blockDim.x;
    for (; i < n; i += stride) p[i] = v;
}

__global__ void deg_accum(const int* __restrict__ dst, float* __restrict__ deg, int E) {
    int e = blockIdx.x * blockDim.x + threadIdx.x;
    if (e < E) atomicAdd(&deg[dst[e]], 1.0f);
}

__global__ void rsqrt_inplace(float* __restrict__ d, int n) {
    int i = blockIdx.x * blockDim.x + threadIdx.x;
    if (i < n) d[i] = rsqrtf(d[i]);
}

__global__ void f32_to_f16(const float* __restrict__ in, _Float16* __restrict__ out, long long n) {
    long long i = (long long)blockIdx.x * blockDim.x + threadIdx.x;
    long long stride = (long long)gridDim.x * blockDim.x;
    for (; i < n; i += stride) out[i] = (_Float16)in[i];
}

// Wt[n*K + k] = W[k*N + n]   (f32 [K,N] -> f16 [N,K])
__global__ void transpose_to_f16(const float* __restrict__ W, _Float16* __restrict__ Wt,
                                 int K, int N) {
    int idx = blockIdx.x * blockDim.x + threadIdx.x;
    if (idx >= K * N) return;
    int k = idx / N, n = idx % N;
    Wt[(long long)n * K + k] = (_Float16)W[(long long)k * N + n];
}

// out[n*d + k] = (f16) x[perm[n]*d + k]
__global__ void perm_convert(const float* __restrict__ x, const int* __restrict__ perm,
                             _Float16* __restrict__ out, int n, int d) {
    long long i = (long long)blockIdx.x * blockDim.x + threadIdx.x;
    if (i >= (long long)n * d) return;
    int row = (int)(i / d), col = (int)(i % d);
    out[i] = (_Float16)x[(long long)perm[row] * d + col];
}

// ============================================================
// WMMA GEMM: C[M,N](f32) = A[M,K](f16, row-major) @ BT[N,K](f16, row-major)^T
// grid = (M/16, N/128), block = 256 (8 waves); wave -> one 16x16 tile
// ============================================================
__global__ __launch_bounds__(256)
void gemm_wmma_f16(const _Float16* __restrict__ A, const _Float16* __restrict__ BT,
                   float* __restrict__ C, int N, int K) {
    const int wave = threadIdx.x >> 5;
    const int lane = threadIdx.x & 31;
    const int mTile = blockIdx.x;
    const int nTile = blockIdx.y * 8 + wave;
    const int half = lane & 15;          // row (A) / col (B) within tile
    const int g    = lane >> 4;          // K-half selector per ISA layout

    const _Float16* arow = A  + (long long)(mTile * 16 + half) * K;
    const _Float16* brow = BT + (long long)(nTile * 16 + half) * K;

    v8f acc = {};
    for (int k0 = 0; k0 < K; k0 += 32) {
        v8h alo = *(const v8h*)(arow + k0 + g * 8);        // K = k0 + g*8 + 0..7
        v8h ahi = *(const v8h*)(arow + k0 + 16 + g * 8);   // K = k0 + 16 + g*8 + 0..7
        v8h blo = *(const v8h*)(brow + k0 + g * 8);
        v8h bhi = *(const v8h*)(brow + k0 + 16 + g * 8);
        v16h a, b;
#pragma unroll
        for (int i = 0; i < 8; ++i) {
            a[i] = alo[i]; a[i + 8] = ahi[i];
            b[i] = blo[i]; b[i + 8] = bhi[i];
        }
        acc = __builtin_amdgcn_wmma_f32_16x16x32_f16(
            /*neg_a=*/false, a, /*neg_b=*/false, b,
            /*c_mod=*/(short)0, acc, /*reuse_a=*/false, /*reuse_b=*/false);
    }
    // D layout: VGPR i -> row g*8+i, col = lane&15
    const int col   = nTile * 16 + half;
    const int rbase = mTile * 16 + g * 8;
#pragma unroll
    for (int i = 0; i < 8; ++i)
        C[(long long)(rbase + i) * N + col] = acc[i];
}

// ============================================================
// edge scatter: agg[dst] += h[src] * dis[src]*dis[dst]   (one block per edge)
// ============================================================
__global__ __launch_bounds__(256)
void edge_scatter(const float* __restrict__ h, const int* __restrict__ src,
                  const int* __restrict__ dst, const float* __restrict__ dis,
                  float* __restrict__ agg) {
    __shared__ float norm;
    __shared__ int s, d;
    if (threadIdx.x == 0) {
        s = src[blockIdx.x]; d = dst[blockIdx.x];
        norm = dis[s] * dis[d];
    }
    __syncthreads();
    int f = threadIdx.x;
    atomicAdd(&agg[(long long)d * DH + f], h[(long long)s * DH + f] * norm);
}

// combine layer1: out_h16 = relu(agg + dis^2*hpre + b)
__global__ __launch_bounds__(256)
void combine_relu_f16(const float* __restrict__ agg, const float* __restrict__ hpre,
                      const float* __restrict__ dis, const float* __restrict__ b,
                      _Float16* __restrict__ out) {
    long long i = (long long)blockIdx.x * DH + threadIdx.x;
    float d = dis[blockIdx.x];
    float v = agg[i] + d * d * hpre[i] + b[threadIdx.x];
    out[i] = (_Float16)(v > 0.0f ? v : 0.0f);
}

// combine layer2: out_f32 = agg + dis^2*hpre + b
__global__ __launch_bounds__(256)
void combine_f32(const float* __restrict__ agg, const float* __restrict__ hpre,
                 const float* __restrict__ dis, const float* __restrict__ b,
                 float* __restrict__ out) {
    long long i = (long long)blockIdx.x * DH + threadIdx.x;
    float d = dis[blockIdx.x];
    out[i] = agg[i] + d * d * hpre[i] + b[threadIdx.x];
}

// readout atomics: sums[batch[n]] += h[n]; cnts[batch[n]] += 1
__global__ __launch_bounds__(256)
void readout(const float* __restrict__ h, const int* __restrict__ batch,
             float* __restrict__ sums, float* __restrict__ cnts) {
    int n = blockIdx.x, g = batch[n], f = threadIdx.x;
    atomicAdd(&sums[(long long)g * DH + f], h[(long long)n * DH + f]);
    if (f == 0) atomicAdd(&cnts[g], 1.0f);
}

__global__ void sigmoid_mean(const float* __restrict__ sums, const float* __restrict__ cnts,
                             float* __restrict__ c) {
    int i = blockIdx.x * blockDim.x + threadIdx.x;
    if (i >= NG * DH) return;
    float m = sums[i] / fmaxf(cnts[i >> 8], 1.0f);
    c[i] = 1.0f / (1.0f + expf(-m));
}

// T64[g][i] = sum_j c[g][j] * Wb[i][j]   (c @ Wb^T, f32, tiny)
__global__ __launch_bounds__(256)
void bilinear_t64(const float* __restrict__ c, const float* __restrict__ Wb,
                  float* __restrict__ T) {
    int g = blockIdx.x, i = threadIdx.x;
    const float* cg = c + (long long)g * DH;
    const float* wr = Wb + (long long)i * DH;
    float acc = 0.0f;
    for (int j = 0; j < DH; ++j) acc += cg[j] * wr[j];
    T[(long long)g * DH + i] = acc;
}

// out[n] = dot(h[n], T[batch[n]]) + bb  (one wave per node, wave32 reduce)
__global__ __launch_bounds__(256)
void score_kernel(const float* __restrict__ h, const float* __restrict__ T,
                  const int* __restrict__ batch, const float* __restrict__ bb,
                  float* __restrict__ out, int nNodes) {
    int wave = threadIdx.x >> 5, lane = threadIdx.x & 31;
    int node = blockIdx.x * 8 + wave;
    if (node >= nNodes) return;
    const float* hr = h + (long long)node * DH;
    const float* tr = T + (long long)batch[node] * DH;
    float acc = 0.0f;
    for (int j = lane; j < DH; j += 32) acc += hr[j] * tr[j];
#pragma unroll
    for (int m = 16; m > 0; m >>= 1) acc += __shfl_xor(acc, m, 32);
    if (lane == 0) out[node] = acc + bb[0];
}

// ============================================================
// host side
// ============================================================
extern "C" void kernel_launch(void* const* d_in, const int* in_sizes, int n_in,
                              void* d_out, int out_size, void* d_ws, size_t ws_size,
                              hipStream_t stream) {
    (void)in_sizes; (void)n_in; (void)out_size; (void)ws_size;
    const float* x    = (const float*)d_in[0];
    const float* W1   = (const float*)d_in[1];
    const float* b1   = (const float*)d_in[2];
    const float* W2   = (const float*)d_in[3];
    const float* b2   = (const float*)d_in[4];
    const float* Wb   = (const float*)d_in[5];
    const float* bb   = (const float*)d_in[6];
    const int*   src  = (const int*)d_in[7];
    const int*   dst  = (const int*)d_in[8];
    const int*   batch= (const int*)d_in[9];
    const int*   perm = (const int*)d_in[10];

    float* logits = (float*)d_out;                 // [2*NN]
    float* h1_out = (float*)d_out + 2 * NN;        // [NN, DH]

    // ---- workspace carve-out (256B aligned) ----
    char* ws = (char*)d_ws;
    size_t off = 0;
    auto carve = [&](size_t bytes) { char* p = ws + off; off = (off + bytes + 255) & ~(size_t)255; return p; };
    float*    dis   = (float*)   carve((size_t)NN * 4);
    _Float16* xh    = (_Float16*)carve((size_t)NN * DIN * 2);
    _Float16* W1t   = (_Float16*)carve((size_t)DH * DIN * 2);
    _Float16* W2t   = (_Float16*)carve((size_t)DH * DH * 2);
    float*    Hpre  = (float*)   carve((size_t)NN * DH * 4);
    float*    AGG   = (float*)   carve((size_t)NN * DH * 4);
    _Float16* Hl1h  = (_Float16*)carve((size_t)NN * DH * 2);
    float*    h2buf = (float*)   carve((size_t)NN * DH * 4);
    float*    sums  = (float*)   carve((size_t)NG * DH * 4);
    float*    cnts  = (float*)   carve((size_t)NG * 4);
    float*    cbuf  = (float*)   carve((size_t)NG * DH * 4);
    float*    T64   = (float*)   carve((size_t)NG * DH * 4);

    const long long ND = (long long)NN * DH;

    // ---- degrees -> dis = (deg+1)^-1/2 ----
    fill_f32<<<256, 256, 0, stream>>>(dis, 1.0f, NN);
    deg_accum<<<(NE + 255) / 256, 256, 0, stream>>>(dst, dis, NE);
    rsqrt_inplace<<<(NN + 255) / 256, 256, 0, stream>>>(dis, NN);

    // ---- f16 conversions ----
    f32_to_f16<<<4096, 256, 0, stream>>>(x, xh, (long long)NN * DIN);
    transpose_to_f16<<<(DIN * DH + 255) / 256, 256, 0, stream>>>(W1, W1t, DIN, DH);
    transpose_to_f16<<<(DH * DH + 255) / 256, 256, 0, stream>>>(W2, W2t, DH, DH);

    dim3 gemmGrid(NN / 16, DH / 128);  // 3125 x 2

    // ================= positive pass =================
    gemm_wmma_f16<<<gemmGrid, 256, 0, stream>>>(xh, W1t, Hpre, DH, DIN);
    fill_f32<<<4096, 256, 0, stream>>>(AGG, 0.0f, ND);
    edge_scatter<<<NE, 256, 0, stream>>>(Hpre, src, dst, dis, AGG);
    combine_relu_f16<<<NN, 256, 0, stream>>>(AGG, Hpre, dis, b1, Hl1h);

    gemm_wmma_f16<<<gemmGrid, 256, 0, stream>>>(Hl1h, W2t, Hpre, DH, DH);
    fill_f32<<<4096, 256, 0, stream>>>(AGG, 0.0f, ND);
    edge_scatter<<<NE, 256, 0, stream>>>(Hpre, src, dst, dis, AGG);
    combine_f32<<<NN, 256, 0, stream>>>(AGG, Hpre, dis, b2, h1_out);

    // ---- readout: c = sigmoid(mean per graph) ----
    fill_f32<<<64, 256, 0, stream>>>(sums, 0.0f, (long long)NG * DH);
    fill_f32<<<1, 64, 0, stream>>>(cnts, 0.0f, NG);
    readout<<<NN, 256, 0, stream>>>(h1_out, batch, sums, cnts);
    sigmoid_mean<<<(NG * DH + 255) / 256, 256, 0, stream>>>(sums, cnts, cbuf);

    // ================= negative pass (shuffled features) =================
    perm_convert<<<(int)(((long long)NN * DIN + 255) / 256), 256, 0, stream>>>(x, perm, xh, NN, DIN);
    gemm_wmma_f16<<<gemmGrid, 256, 0, stream>>>(xh, W1t, Hpre, DH, DIN);
    fill_f32<<<4096, 256, 0, stream>>>(AGG, 0.0f, ND);
    edge_scatter<<<NE, 256, 0, stream>>>(Hpre, src, dst, dis, AGG);
    combine_relu_f16<<<NN, 256, 0, stream>>>(AGG, Hpre, dis, b1, Hl1h);

    gemm_wmma_f16<<<gemmGrid, 256, 0, stream>>>(Hl1h, W2t, Hpre, DH, DH);
    fill_f32<<<4096, 256, 0, stream>>>(AGG, 0.0f, ND);
    edge_scatter<<<NE, 256, 0, stream>>>(Hpre, src, dst, dis, AGG);
    combine_f32<<<NN, 256, 0, stream>>>(AGG, Hpre, dis, b2, h2buf);

    // ---- bilinear discriminator ----
    bilinear_t64<<<NG, 256, 0, stream>>>(cbuf, Wb, T64);
    score_kernel<<<NN / 8, 256, 0, stream>>>(h1_out, T64, batch, bb, logits, NN);
    score_kernel<<<NN / 8, 256, 0, stream>>>(h2buf, T64, batch, bb, logits + NN, NN);
}
